// ARDecoder_89120571392491
// MI455X (gfx1250) — matrix operand order repaired and checked by reading, not compiled, not run on hardware
//
#include <hip/hip_runtime.h>
#include <math.h>

// ---------------------------------------------------------------------------
// Problem constants
// ---------------------------------------------------------------------------
#define BB   32
#define TT   1000
#define CEc  256
#define DD   1024
#define PHc  256
#define FRc  160       // COUT*R
#define COUTc 80
#define NWG  32        // persistent workgroups in the scan kernel (64 tiles / 2)

typedef __attribute__((ext_vector_type(16))) __bf16 v16bf;
typedef __attribute__((ext_vector_type(8)))  __bf16 v8bf;
typedef __attribute__((ext_vector_type(8)))  float  v8f;

// ---------------------------------------------------------------------------
// WMMA fragment helpers (CDNA5 16x16x32 bf16, wave32)
//
// A-matrix 16x32 (MxK), contiguous k order used here (A and B match):
//   lanes 0-15 : M=lane,    elems 0..15 = K 0..15
//   lanes16-31 : M=lane-16, elems 0..15 = K 16..31
// B-matrix 32x16 (KxN) loaded from row-major W[N x K] (i.e. B = W^T):
//   lanes 0-15 : N=lane,    elems 0..15 = K 0..15
//   lanes16-31 : N=lane-16, elems 0..15 = K 16..31
// ---------------------------------------------------------------------------
__device__ __forceinline__ v16bf load_a_frag(const __bf16* tile, size_t lda, int lane) {
  const int m = lane & 15, sel = lane >> 4;
  const __bf16* p = tile + (size_t)m * lda + (size_t)(sel * 16);
  v16bf r;
  ((v8bf*)&r)[0] = *(const v8bf*)(p);
  ((v8bf*)&r)[1] = *(const v8bf*)(p + 8);
  return r;
}

__device__ __forceinline__ v16bf load_b_frag(const __bf16* wtile, size_t ldb, int lane) {
  const int n = lane & 15, sel = lane >> 4;
  const __bf16* p = wtile + (size_t)n * ldb + (size_t)(sel * 16);
  v16bf r;
  ((v8bf*)&r)[0] = *(const v8bf*)(p);
  ((v8bf*)&r)[1] = *(const v8bf*)(p + 8);
  return r;
}

__device__ __forceinline__ v8f wmma_bf16(v16bf a, v16bf b, v8f c) {
  return __builtin_amdgcn_wmma_f32_16x16x32_bf16(false, a, false, b, (short)0, c,
                                                 false, false);
}

__device__ __forceinline__ float sigmoidf_(float x) { return 1.0f / (1.0f + expf(-x)); }

// Opaque value pass-through: volatile asm inside the time loop cannot be
// hoisted, so loads whose addresses depend on the result cannot be LICM'd
// out of the 1000-iteration loop (prevents the compiler from caching 100s
// of loop-invariant weight fragments -> scratch spills / VGPR-MSB pressure).
// We launder the OFFSET (not the pointer) so address-space inference on the
// kernel-arg base pointer survives and loads stay global_load (not flat_load,
// which would also consume DScnt and contend with LDS traffic).
__device__ __forceinline__ size_t opaque_off(size_t x) {
  asm volatile("" : "+v"(x));
  return x;
}

// ---------------------------------------------------------------------------
// Small prep kernels
// ---------------------------------------------------------------------------
__global__ void cvt_f32_bf16(const float* __restrict__ in, __bf16* __restrict__ out, int n) {
  int i = blockIdx.x * blockDim.x + threadIdx.x;
  if (i < n) out[i] = (__bf16)in[i];
}

__global__ void add_vec(const float* __restrict__ a, const float* __restrict__ b,
                        float* __restrict__ out, int n) {
  int i = blockIdx.x * blockDim.x + threadIdx.x;
  if (i < n) out[i] = a[i] + b[i];
}

// prev frames: (B,T,160) bf16; row b*T+t; f = r*80+c; value = ftgt[b,c,(t-1)*2+r], t>0
__global__ void build_prev(const float* __restrict__ ftgt, __bf16* __restrict__ prevb) {
  int i = blockIdx.x * blockDim.x + threadIdx.x;
  if (i >= BB * TT * FRc) return;
  int f = i % FRc; int row = i / FRc;
  int t = row % TT; int b = row / TT;
  float v = 0.0f;
  if (t > 0) {
    int c = f % COUTc, r = f / COUTc;
    v = ftgt[((size_t)b * COUTc + c) * (2 * TT) + (size_t)(t - 1) * 2 + r];
  }
  prevb[i] = (__bf16)v;
}

// enc -> xcat cols [0,256) and hcat cols [1024,1280)
__global__ void build_enc(const float* __restrict__ enc, __bf16* __restrict__ xcat,
                          __bf16* __restrict__ hcat) {
  int i = blockIdx.x * blockDim.x + threadIdx.x;
  if (i >= BB * TT * CEc) return;
  int k = i % CEc; int row = i / CEc;
  float v = enc[i];
  xcat[(size_t)row * 512  + k]        = (__bf16)v;
  hcat[(size_t)row * 1280 + 1024 + k] = (__bf16)v;
}

// final permute: fout (B*T,160) -> out (B,80,2T)
__global__ void out_permute(const float* __restrict__ fout, float* __restrict__ out) {
  int i = blockIdx.x * blockDim.x + threadIdx.x;
  if (i >= BB * TT * FRc) return;
  int f = i % FRc; int row = i / FRc;
  int t = row % TT; int b = row / TT;
  int c = f % COUTc, r = f / COUTc;
  out[((size_t)b * COUTc + c) * (2 * TT) + (size_t)t * 2 + r] = fout[i];
}

// ---------------------------------------------------------------------------
// Generic time-parallel WMMA GEMM: C = act(A[MxK]bf16 @ W[NxK]bf16^T + bias)
// One wave computes a 32x32 supertile (2x2 WMMA tiles, shared A/B frags).
// M,N multiples of 32; K multiple of 32.
// ---------------------------------------------------------------------------
template <bool OBF16, bool RELU>
__global__ __launch_bounds__(256) void wmma_gemm(
    const __bf16* __restrict__ A, size_t lda,
    const __bf16* __restrict__ W, size_t ldb,
    const float* __restrict__ bias,
    void* __restrict__ Cout, size_t ldc,
    int M, int N, int K) {
  const int lane = threadIdx.x & 31;
  const int wave = blockIdx.x * (blockDim.x >> 5) + (threadIdx.x >> 5);
  const int nSup = N >> 5;
  const int total = (M >> 5) * nSup;
  if (wave >= total) return;             // whole-wave uniform: EXEC stays all-1s
  const int sm = wave / nSup, sn = wave % nSup;
  const size_t m0 = (size_t)sm * 32;
  const int n0 = sn * 32;
  const int cl = lane & 15, sel = lane >> 4;

  const float b0 = bias ? bias[n0 + cl] : 0.0f;
  const float b1 = bias ? bias[n0 + 16 + cl] : 0.0f;
  v8f acc00, acc01, acc10, acc11;
#pragma unroll
  for (int v = 0; v < 8; ++v) { acc00[v] = b0; acc01[v] = b1; acc10[v] = b0; acc11[v] = b1; }

  const __bf16* a0p = A + m0 * lda;
  const __bf16* a1p = A + (m0 + 16) * lda;
  const __bf16* w0p = W + (size_t)n0 * ldb;
  const __bf16* w1p = W + (size_t)(n0 + 16) * ldb;

#pragma clang loop unroll_count(4)
  for (int kb = 0; kb < K; kb += 32) {
    v16bf a0 = load_a_frag(a0p + kb, lda, lane);
    v16bf a1 = load_a_frag(a1p + kb, lda, lane);
    v16bf w0 = load_b_frag(w0p + kb, ldb, lane);
    v16bf w1 = load_b_frag(w1p + kb, ldb, lane);
    acc00 = wmma_bf16(a0, w0, acc00);
    acc01 = wmma_bf16(a0, w1, acc01);
    acc10 = wmma_bf16(a1, w0, acc10);
    acc11 = wmma_bf16(a1, w1, acc11);
  }

#pragma unroll
  for (int v = 0; v < 8; ++v) {
    const size_t r0 = m0 + v + 8 * sel;
    const size_t r1 = r0 + 16;
    float v00 = acc00[v], v01 = acc01[v], v10 = acc10[v], v11 = acc11[v];
    if (RELU) {
      v00 = v00 > 0.f ? v00 : 0.f; v01 = v01 > 0.f ? v01 : 0.f;
      v10 = v10 > 0.f ? v10 : 0.f; v11 = v11 > 0.f ? v11 : 0.f;
    }
    if (OBF16) {
      __bf16* C = (__bf16*)Cout;
      C[r0 * ldc + n0 + cl]      = (__bf16)v00;
      C[r0 * ldc + n0 + 16 + cl] = (__bf16)v01;
      C[r1 * ldc + n0 + cl]      = (__bf16)v10;
      C[r1 * ldc + n0 + 16 + cl] = (__bf16)v11;
    } else {
      float* C = (float*)Cout;
      C[r0 * ldc + n0 + cl]      = v00;
      C[r0 * ldc + n0 + 16 + cl] = v01;
      C[r1 * ldc + n0 + cl]      = v10;
      C[r1 * ldc + n0 + 16 + cl] = v11;
    }
  }
}

// ---------------------------------------------------------------------------
// Device-wide barrier for the persistent scan kernel (arrive counter + gen).
// ---------------------------------------------------------------------------
__device__ __forceinline__ void grid_barrier(int* bar) {
  __syncthreads();
  if (threadIdx.x == 0) {
    __threadfence();
    int gen = atomicAdd(bar + 1, 0);
    int prev = atomicAdd(bar, 1);
    if (prev == NWG - 1) {
      atomicExch(bar, 0);
      __threadfence();
      atomicAdd(bar + 1, 1);
    } else {
      while (atomicAdd(bar + 1, 0) == gen) { __builtin_amdgcn_s_sleep(2); }
    }
    __threadfence();
  }
  __syncthreads();
}

// ---------------------------------------------------------------------------
// Persistent scan kernel: 32 workgroups x 256 threads (8 waves).
// WG w owns hidden columns [w*32, w*32+32) of BOTH LSTM layers.
// Wave layout per GEMM: wave = gate*2 + htile; computes two 16x16 M-tiles
// (batch 0..15 / 16..31) of the (gate*1024 + slice) output columns.
// gx0 is folded in (extra K=512 over xcat). h-states double-buffered in
// global bf16 so all WGs see the full 1024-wide h as the next GEMM's A.
// Two grid barriers per step; c-state and zoneout h-state stay in LDS.
// Weight fragments stream from L2 every step (28 MB/step << 192 MB L2);
// opaque_off() keeps the compiler from hoisting+spilling them.
// ---------------------------------------------------------------------------
__global__ __launch_bounds__(256, 1) void scan_kernel(
    const __bf16* __restrict__ xcat,   // (B*T, 512) row = b*T+t
    const __bf16* __restrict__ wih0b,  // (4096, 512)
    const __bf16* __restrict__ whh0b,  // (4096, 1024)
    const __bf16* __restrict__ wih1b,  // (4096, 1024)
    const __bf16* __restrict__ whh1b,  // (4096, 1024)
    const float*  __restrict__ bsum0,  // 4096 (bih0+bhh0)
    const float*  __restrict__ bsum1,  // 4096 (bih1+bhh1)
    __bf16* __restrict__ h0g,          // [2][32][1024] double-buffered
    __bf16* __restrict__ h1g,          // [2][32][1024]
    __bf16* __restrict__ hcat,         // (B*T, 1280), cols [0,1024) written here
    int* __restrict__ bar) {
  __shared__ float gbuf[4][32][32];
  __shared__ float c0s[32][32], h0s[32][32], c1s[32][32], h1s[32][32];

  const int tid = threadIdx.x;
  const int lane = tid & 31;
  const int wave = tid >> 5;
  const int wg = blockIdx.x;
  const int colbase = wg * 32;
  const int gate = wave >> 1;
  const int ht = wave & 1;
  const int n0 = gate * 1024 + colbase + ht * 16;
  const int cl = lane & 15, sel = lane >> 4;

  for (int i = tid; i < 32 * 32; i += 256) {
    int r = i >> 5, c = i & 31;
    c0s[r][c] = 0.f; h0s[r][c] = 0.f; c1s[r][c] = 0.f; h1s[r][c] = 0.f;
  }
  __syncthreads();

  const float bias0 = bsum0[n0 + cl];
  const float bias1 = bsum1[n0 + cl];
  const size_t XLDA = (size_t)TT * 512;   // row stride between batches in xcat

  for (int t = 0; t < TT; ++t) {
    const int p = t & 1;
    const __bf16* h0prev = h0g + (size_t)p * (32 * 1024);
    __bf16*       h0next = h0g + (size_t)(p ^ 1) * (32 * 1024);
    const __bf16* h1prev = h1g + (size_t)p * (32 * 1024);
    __bf16*       h1next = h1g + (size_t)(p ^ 1) * (32 * 1024);

    // Opaque per-iteration offsets: defeats cross-t LICM of weight loads
    // while keeping the kernel-arg base (=> global_load, not flat_load).
    const __bf16* whh0r = whh0b + opaque_off((size_t)n0 * 1024);
    const __bf16* wih0r = wih0b + opaque_off((size_t)n0 * 512);

    // ---------------- layer 0: g0 = bsum0 + x_t@wih0^T + h0@whh0^T --------
    v8f acc0, acc1;
#pragma unroll
    for (int v = 0; v < 8; ++v) { acc0[v] = bias0; acc1[v] = bias0; }

#pragma clang loop unroll_count(4)
    for (int kb = 0; kb < 1024; kb += 32) {
      v16bf b  = load_b_frag(whh0r + kb, 1024, lane);
      v16bf a0 = load_a_frag(h0prev + kb, 1024, lane);
      v16bf a1 = load_a_frag(h0prev + 16 * 1024 + kb, 1024, lane);
      acc0 = wmma_bf16(a0, b, acc0);
      acc1 = wmma_bf16(a1, b, acc1);
    }
    const __bf16* xbase = xcat + (size_t)t * 512;
#pragma clang loop unroll_count(4)
    for (int kb = 0; kb < 512; kb += 32) {
      v16bf b  = load_b_frag(wih0r + kb, 512, lane);
      v16bf a0 = load_a_frag(xbase + kb, XLDA, lane);
      v16bf a1 = load_a_frag(xbase + (size_t)16 * XLDA + kb, XLDA, lane);
      acc0 = wmma_bf16(a0, b, acc0);
      acc1 = wmma_bf16(a1, b, acc1);
    }
#pragma unroll
    for (int v = 0; v < 8; ++v) {
      gbuf[gate][v + 8 * sel][ht * 16 + cl]      = acc0[v];
      gbuf[gate][16 + v + 8 * sel][ht * 16 + cl] = acc1[v];
    }
    __syncthreads();

    // elementwise LSTM0 + zoneout; publish new h0 (bf16) globally
#pragma unroll
    for (int j = 0; j < 4; ++j) {
      int cell = tid + 256 * j;
      int r = cell >> 5, c = cell & 31;
      float gi = gbuf[0][r][c], gf = gbuf[1][r][c];
      float gg = gbuf[2][r][c], go = gbuf[3][r][c];
      float cold = c0s[r][c], hold = h0s[r][c];
      float cn = sigmoidf_(gf) * cold + sigmoidf_(gi) * tanhf(gg);
      float hn = sigmoidf_(go) * tanhf(cn);
      float cm = 0.1f * cold + 0.9f * cn;
      float hm = 0.1f * hold + 0.9f * hn;
      c0s[r][c] = cm; h0s[r][c] = hm;
      h0next[(size_t)r * 1024 + colbase + c] = (__bf16)hm;
    }
    __threadfence();
    grid_barrier(bar);

    // ---------------- layer 1: g1 = bsum1 + h0@wih1^T + h1@whh1^T ---------
    const __bf16* wih1r = wih1b + opaque_off((size_t)n0 * 1024);
    const __bf16* whh1r = whh1b + opaque_off((size_t)n0 * 1024);

#pragma unroll
    for (int v = 0; v < 8; ++v) { acc0[v] = bias1; acc1[v] = bias1; }

#pragma clang loop unroll_count(4)
    for (int kb = 0; kb < 1024; kb += 32) {
      v16bf b  = load_b_frag(wih1r + kb, 1024, lane);
      v16bf a0 = load_a_frag(h0next + kb, 1024, lane);
      v16bf a1 = load_a_frag(h0next + 16 * 1024 + kb, 1024, lane);
      acc0 = wmma_bf16(a0, b, acc0);
      acc1 = wmma_bf16(a1, b, acc1);
    }
#pragma clang loop unroll_count(4)
    for (int kb = 0; kb < 1024; kb += 32) {
      v16bf b  = load_b_frag(whh1r + kb, 1024, lane);
      v16bf a0 = load_a_frag(h1prev + kb, 1024, lane);
      v16bf a1 = load_a_frag(h1prev + 16 * 1024 + kb, 1024, lane);
      acc0 = wmma_bf16(a0, b, acc0);
      acc1 = wmma_bf16(a1, b, acc1);
    }
#pragma unroll
    for (int v = 0; v < 8; ++v) {
      gbuf[gate][v + 8 * sel][ht * 16 + cl]      = acc0[v];
      gbuf[gate][16 + v + 8 * sel][ht * 16 + cl] = acc1[v];
    }
    __syncthreads();

#pragma unroll
    for (int j = 0; j < 4; ++j) {
      int cell = tid + 256 * j;
      int r = cell >> 5, c = cell & 31;
      float gi = gbuf[0][r][c], gf = gbuf[1][r][c];
      float gg = gbuf[2][r][c], go = gbuf[3][r][c];
      float cold = c1s[r][c], hold = h1s[r][c];
      float cn = sigmoidf_(gf) * cold + sigmoidf_(gi) * tanhf(gg);
      float hn = sigmoidf_(go) * tanhf(cn);
      float cm = 0.1f * cold + 0.9f * cn;
      float hm = 0.1f * hold + 0.9f * hn;
      c1s[r][c] = cm; h1s[r][c] = hm;
      h1next[(size_t)r * 1024 + colbase + c] = (__bf16)hm;
      hcat[((size_t)r * TT + t) * 1280 + colbase + c] = (__bf16)hm;
    }
    __threadfence();
    grid_barrier(bar);
  }
}

// ---------------------------------------------------------------------------
// Host launcher
// ---------------------------------------------------------------------------
extern "C" void kernel_launch(void* const* d_in, const int* in_sizes, int n_in,
                              void* d_out, int out_size, void* d_ws, size_t ws_size,
                              hipStream_t stream) {
  (void)in_sizes; (void)n_in; (void)out_size; (void)ws_size;
  const float* enc  = (const float*)d_in[0];
  const float* ftgt = (const float*)d_in[1];
  const float* pw1  = (const float*)d_in[2];
  const float* pb1  = (const float*)d_in[3];
  const float* pw2  = (const float*)d_in[4];
  const float* pb2  = (const float*)d_in[5];
  const float* wih0 = (const float*)d_in[6];
  const float* whh0 = (const float*)d_in[7];
  const float* bih0 = (const float*)d_in[8];
  const float* bhh0 = (const float*)d_in[9];
  const float* wih1 = (const float*)d_in[10];
  const float* whh1 = (const float*)d_in[11];
  const float* bih1 = (const float*)d_in[12];
  const float* bhh1 = (const float*)d_in[13];
  const float* wfo  = (const float*)d_in[14];
  float* out = (float*)d_out;

  char* wsp = (char*)d_ws;
  auto alloc = [&](size_t bytes) -> char* {
    char* p = wsp;
    wsp += (bytes + 255) & ~(size_t)255;
    return p;
  };
  __bf16* wih0b = (__bf16*)alloc((size_t)4096 * 512 * 2);
  __bf16* whh0b = (__bf16*)alloc((size_t)4096 * 1024 * 2);
  __bf16* wih1b = (__bf16*)alloc((size_t)4096 * 1024 * 2);
  __bf16* whh1b = (__bf16*)alloc((size_t)4096 * 1024 * 2);
  __bf16* pw1b  = (__bf16*)alloc((size_t)PHc * FRc * 2);
  __bf16* pw2b  = (__bf16*)alloc((size_t)PHc * PHc * 2);
  __bf16* wfob  = (__bf16*)alloc((size_t)FRc * 1280 * 2);
  float*  bsum0 = (float*)alloc(4096 * 4);
  float*  bsum1 = (float*)alloc(4096 * 4);
  __bf16* prevb = (__bf16*)alloc((size_t)BB * TT * FRc * 2);
  __bf16* pre1b = (__bf16*)alloc((size_t)BB * TT * PHc * 2);
  __bf16* xcatb = (__bf16*)alloc((size_t)BB * TT * 512 * 2);
  __bf16* hcatb = (__bf16*)alloc((size_t)BB * TT * 1280 * 2);
  __bf16* h0g   = (__bf16*)alloc((size_t)2 * 32 * 1024 * 2);
  __bf16* h1g   = (__bf16*)alloc((size_t)2 * 32 * 1024 * 2);
  float*  foutb = (float*)alloc((size_t)BB * TT * FRc * 4);
  int*    bar   = (int*)alloc(256);

  auto cdiv = [](long a, long b) { return (int)((a + b - 1) / b); };

  // 1) weights -> bf16
  cvt_f32_bf16<<<cdiv(4096L * 512, 256), 256, 0, stream>>>(wih0, wih0b, 4096 * 512);
  cvt_f32_bf16<<<cdiv(4096L * 1024, 256), 256, 0, stream>>>(whh0, whh0b, 4096 * 1024);
  cvt_f32_bf16<<<cdiv(4096L * 1024, 256), 256, 0, stream>>>(wih1, wih1b, 4096 * 1024);
  cvt_f32_bf16<<<cdiv(4096L * 1024, 256), 256, 0, stream>>>(whh1, whh1b, 4096 * 1024);
  cvt_f32_bf16<<<cdiv((long)PHc * FRc, 256), 256, 0, stream>>>(pw1, pw1b, PHc * FRc);
  cvt_f32_bf16<<<cdiv((long)PHc * PHc, 256), 256, 0, stream>>>(pw2, pw2b, PHc * PHc);
  cvt_f32_bf16<<<cdiv((long)FRc * 1280, 256), 256, 0, stream>>>(wfo, wfob, FRc * 1280);
  // 2) fused biases
  add_vec<<<16, 256, 0, stream>>>(bih0, bhh0, bsum0, 4096);
  add_vec<<<16, 256, 0, stream>>>(bih1, bhh1, bsum1, 4096);
  // 3) teacher-forced prev frames + enc copies
  build_prev<<<cdiv((long)BB * TT * FRc, 256), 256, 0, stream>>>(ftgt, prevb);
  build_enc<<<cdiv((long)BB * TT * CEc, 256), 256, 0, stream>>>(enc, xcatb, hcatb);

  // 4) prenet layer 1: pre1 = relu(prev @ pw1^T + pb1), (32000 x 256) bf16
  {
    int waves = (BB * TT / 32) * (PHc / 32);
    wmma_gemm<true, true><<<cdiv(waves, 8), 256, 0, stream>>>(
        prevb, (size_t)FRc, pw1b, (size_t)FRc, pb1, (void*)pre1b, (size_t)PHc,
        BB * TT, PHc, FRc);
  }
  // 5) prenet layer 2 -> xcat cols [256,512): relu(pre1 @ pw2^T + pb2)
  {
    int waves = (BB * TT / 32) * (PHc / 32);
    wmma_gemm<true, true><<<cdiv(waves, 8), 256, 0, stream>>>(
        pre1b, (size_t)PHc, pw2b, (size_t)PHc, pb2, (void*)(xcatb + 256), (size_t)512,
        BB * TT, PHc, PHc);
  }

  // 6) zero h-state double buffers and the device barrier
  (void)hipMemsetAsync(h0g, 0, (size_t)2 * 32 * 1024 * 2, stream);
  (void)hipMemsetAsync(h1g, 0, (size_t)2 * 32 * 1024 * 2, stream);
  (void)hipMemsetAsync(bar, 0, 256, stream);

  // 7) persistent AR scan (1000 steps, 2 grid barriers/step)
  scan_kernel<<<NWG, 256, 0, stream>>>(xcatb, wih0b, whh0b, wih1b, whh1b,
                                       bsum0, bsum1, h0g, h1g, hcatb, bar);

  // 8) output projection: fout = hcat @ wfo^T  (32000 x 160), f32
  {
    int waves = (BB * TT / 32) * (FRc / 32);
    wmma_gemm<false, false><<<cdiv(waves, 8), 256, 0, stream>>>(
        hcatb, (size_t)1280, wfob, (size_t)1280, (const float*)nullptr,
        (void*)foutb, (size_t)FRc, BB * TT, FRc, 1280);
  }
  // 9) reshape/permute to (B, 80, 2T)
  out_permute<<<cdiv((long)BB * TT * FRc, 256), 256, 0, stream>>>(foutb, out);
}
